// Deform_conv_27882927685928
// MI455X (gfx1250) — matrix operand order repaired and checked by reading, not compiled, not run on hardware
//
#include <hip/hip_runtime.h>
#include <hip/hip_bf16.h>

typedef _Float16 h16;
typedef __attribute__((ext_vector_type(16))) _Float16 v16h;
typedef __attribute__((ext_vector_type(8)))  _Float16 v8h;
typedef __attribute__((ext_vector_type(8)))  float    v8f;
typedef __attribute__((address_space(3)))    h16      lds_h;

#define CB   8
#define CC   256
#define CH   64
#define CW   64
#define CO   256
#define CKK  9
#define KDIM (CC * CKK)          // 2304, K ordering: k = kk*256 + c
#define HW   (CH * CW)           // 4096
#define BN_EPS 1e-5f

// ---- workspace layout (bytes) ----
#define OFF_SCALE 0u                         // 256 f32
#define OFF_SHIFT 1024u                      // 256 f32
#define OFF_BW    4096u                      // 256*2304 f16 = 1179648
#define OFF_WIDX  1184768u                   // 9*4096 int4  = 589824
#define OFF_WWGT  1774592u                   // 9*4096 float4= 589824
#define OFF_A     2364416u                   // 4096*2304 f16 = 18874368
// total ~21.2 MB

// ------------------------------------------------------------------
// K0: weights -> f16 [O][kk*256+c] (K-permuted); fold bias + BN.
// ------------------------------------------------------------------
__global__ void convert_kernel(const float* __restrict__ w,
                               const float* __restrict__ bias,
                               const float* __restrict__ gamma,
                               const float* __restrict__ beta,
                               const float* __restrict__ rmean,
                               const float* __restrict__ rvar,
                               h16* __restrict__ Bw,
                               float* __restrict__ scale,
                               float* __restrict__ shift) {
    int gid = blockIdx.x * blockDim.x + threadIdx.x;
    if (gid < CO * KDIM) {
        int o  = gid / KDIM;
        int r  = gid - o * KDIM;
        int kk = r >> 8;          // KDIM = 9*256
        int c  = r & 255;
        Bw[gid] = (h16)w[((size_t)o * CC + c) * CKK + kk];
    }
    if (gid < CO) {
        float inv = gamma[gid] * rsqrtf(rvar[gid] + BN_EPS);
        scale[gid] = inv;
        shift[gid] = bias[gid] * inv + beta[gid] - rmean[gid] * inv;
    }
}

// ------------------------------------------------------------------
// K1: offset conv (27ch) + gather metadata for one batch.
// One thread per (h,w); weight reads are wave-uniform (scalarizable).
// ------------------------------------------------------------------
__global__ void __launch_bounds__(256)
offset_meta_kernel(const float* __restrict__ x,
                   const float* __restrict__ w_off,
                   const float* __restrict__ b_off,
                   int b,
                   int4*   __restrict__ widx,
                   float4* __restrict__ wwgt) {
    int pos = blockIdx.x * 256 + threadIdx.x;    // 0..4095
    int h = pos >> 6, w = pos & 63;

    float acc[27];
#pragma unroll
    for (int oc = 0; oc < 27; ++oc) acc[oc] = b_off[oc];

    const float* xb = x + (size_t)b * CC * HW;
    for (int c = 0; c < CC; ++c) {
        const float* xc = xb + (size_t)c * HW;
        float xv[9];
#pragma unroll
        for (int ky = 0; ky < 3; ++ky) {
#pragma unroll
            for (int kx = 0; kx < 3; ++kx) {
                int yy = h + ky - 1, xx = w + kx - 1;
                bool ok = (yy >= 0) & (yy < CH) & (xx >= 0) & (xx < CW);
                xv[ky * 3 + kx] = ok ? xc[yy * CW + xx] : 0.0f;
            }
        }
        const float* wo = w_off + (size_t)c * CKK;
#pragma unroll
        for (int oc = 0; oc < 27; ++oc) {
            const float* wr = wo + (size_t)oc * CC * CKK;
#pragma unroll
            for (int j = 0; j < 9; ++j) acc[oc] += xv[j] * wr[j];
        }
    }

#pragma unroll
    for (int kk = 0; kk < CKK; ++kk) {
        int ky = kk / 3, kx = kk % 3;
        float dy = acc[kk], dx = acc[9 + kk];
        float mod = 1.0f / (1.0f + expf(-acc[18 + kk]));
        float py = (float)(h - 1 + ky) + dy;
        float px = (float)(w - 1 + kx) + dx;
        float fy0 = floorf(py), fx0 = floorf(px);
        int y0 = (int)fy0, x0 = (int)fx0;
        float fy = py - fy0, fx = px - fx0;

        int   id[4];
        float wg[4];
        int   ys[4] = { y0, y0, y0 + 1, y0 + 1 };
        int   xs[4] = { x0, x0 + 1, x0, x0 + 1 };
        float cw[4] = { (1.f - fy) * (1.f - fx), (1.f - fy) * fx,
                        fy * (1.f - fx),          fy * fx };
#pragma unroll
        for (int q = 0; q < 4; ++q) {
            int yq = ys[q], xq = xs[q];
            bool ok = (yq >= 0) & (yq < CH) & (xq >= 0) & (xq < CW);
            int yc  = min(max(yq, 0), CH - 1);
            int xc2 = min(max(xq, 0), CW - 1);
            id[q] = yc * CW + xc2;
            wg[q] = ok ? cw[q] * mod : 0.0f;
        }
        widx[kk * HW + pos] = make_int4(id[0], id[1], id[2], id[3]);
        wwgt[kk * HW + pos] = make_float4(wg[0], wg[1], wg[2], wg[3]);
    }
}

// ------------------------------------------------------------------
// K2: bilinear sample + modulate -> f16 A [4096][2304], k = kk*256+c.
// threadIdx.x = c  ->  all 9 stores are fully coalesced across the wave.
// ------------------------------------------------------------------
__global__ void __launch_bounds__(256)
sample_kernel(const float* __restrict__ x,
              const int4*   __restrict__ widx,
              const float4* __restrict__ wwgt,
              h16* __restrict__ A,
              int b) {
    int pos = blockIdx.x;                 // 0..4095
    int c   = threadIdx.x;                // 0..255
    const float* xc = x + ((size_t)(b * CC + c)) * HW;
    h16* acol = A + (size_t)pos * KDIM + c;
#pragma unroll
    for (int kk = 0; kk < CKK; ++kk) {
        int4   id = widx[kk * HW + pos];
        float4 wg = wwgt[kk * HW + pos];
        float v = wg.x * xc[id.x] + wg.y * xc[id.y] +
                  wg.z * xc[id.z] + wg.w * xc[id.w];
        acol[kk * CC] = (h16)v;
    }
}

// ------------------------------------------------------------------
// K3: WMMA GEMM  C[m,n] = sum_k A[m,k]*B[n,k], M=4096, N=256, K=2304.
// Block = 256 threads (8 waves). Tile 128(M) x 64(N), K-step 64,
// double-buffered LDS filled by GLOBAL_LOAD_ASYNC_TO_LDS_B128 (ASYNCcnt).
// Wave w owns M-subtile w x 4 N-subtiles -> 4 x v8f accumulators.
// ------------------------------------------------------------------
#define ASYNC_B128_0(l, g)  asm volatile("global_load_async_to_lds_b128 %0, %1, off"           :: "v"(l), "v"(g) : "memory")
#define ASYNC_B128_16(l, g) asm volatile("global_load_async_to_lds_b128 %0, %1, off offset:16" :: "v"(l), "v"(g) : "memory")
#define ASYNC_B128_32(l, g) asm volatile("global_load_async_to_lds_b128 %0, %1, off offset:32" :: "v"(l), "v"(g) : "memory")
#define ASYNC_B128_48(l, g) asm volatile("global_load_async_to_lds_b128 %0, %1, off offset:48" :: "v"(l), "v"(g) : "memory")

__global__ void __launch_bounds__(256)
gemm_kernel(const h16* __restrict__ A,
            const h16* __restrict__ Bw,
            const float* __restrict__ scale,
            const float* __restrict__ shift,
            float* __restrict__ out,
            int b) {
    __shared__ h16 As[2][128][72];   // 64 used + 8 pad  (36 KB)
    __shared__ h16 Bs[2][64][72];    //                  (18 KB)

    const int tid  = threadIdx.x;
    const int wave = tid >> 5;
    const int lane = tid & 31;
    const int m_base = blockIdx.x * 128;
    const int n_base = blockIdx.y * 64;

    v8f acc[4];
    v8f zero = {};
#pragma unroll
    for (int nt = 0; nt < 4; ++nt) acc[nt] = zero;

    // tile-fill assignment: A 128x64 halves, B 64x64 halves per K-step
    const int ra = tid >> 1, ca = (tid & 1) * 32;   // 32 halves (4 x b128)
    const int rb = tid >> 2, cb = (tid & 3) * 16;   // 16 halves (2 x b128)

    const h16* gA = A  + (size_t)(m_base + ra) * KDIM + ca;
    const h16* gB = Bw + (size_t)(n_base + rb) * KDIM + cb;

    // per-buffer LDS destinations for this thread's async fills
    lds_h* lA[2] = { (lds_h*)&As[0][ra][ca], (lds_h*)&As[1][ra][ca] };
    lds_h* lB[2] = { (lds_h*)&Bs[0][rb][cb], (lds_h*)&Bs[1][rb][cb] };

    // ISA 7.12.2 wave32 fragment layouts:
    // A 16x32 f16: lanes0-15 K={0..7,16..23}, lanes16-31 K={8..15,24..31}
    // B 32x16 f16: lanes0-15 K={0..7,16..23}, lanes16-31 K={8..15,24..31}, N=lane&15
    const int m_ld   = wave * 16 + (lane & 15);
    const int a_koff = (lane >= 16) ? 8 : 0;
    const int b_koff = (lane >= 16) ? 8 : 0;
    const int n_ld   = lane & 15;

    // prologue: async-fill buffer 0 (6 x b128 per thread)
    ASYNC_B128_0 (lA[0], gA); ASYNC_B128_16(lA[0], gA);
    ASYNC_B128_32(lA[0], gA); ASYNC_B128_48(lA[0], gA);
    ASYNC_B128_0 (lB[0], gB); ASYNC_B128_16(lB[0], gB);
    gA += 64; gB += 64;

    const int NSTEP = KDIM / 64;     // 36
    for (int s = 0; s < NSTEP; ++s) {
        const int buf = s & 1;
        if (s + 1 < NSTEP) {
            const int nb = buf ^ 1;
            ASYNC_B128_0 (lA[nb], gA); ASYNC_B128_16(lA[nb], gA);
            ASYNC_B128_32(lA[nb], gA); ASYNC_B128_48(lA[nb], gA);
            ASYNC_B128_0 (lB[nb], gB); ASYNC_B128_16(lB[nb], gB);
            gA += 64; gB += 64;
            // 6 newest (next tile) may stay in flight; current tile done
            asm volatile("s_wait_asynccnt 0x6" ::: "memory");
        } else {
            asm volatile("s_wait_asynccnt 0x0" ::: "memory");
        }
        __syncthreads();

#pragma unroll
        for (int kc = 0; kc < 64; kc += 32) {
            v8h fal = *(const v8h*)&As[buf][m_ld][kc + a_koff];
            v8h fah = *(const v8h*)&As[buf][m_ld][kc + a_koff + 16];
            v16h afrag;
#pragma unroll
            for (int i = 0; i < 8; ++i) { afrag[i] = fal[i]; afrag[i + 8] = fah[i]; }

#pragma unroll
            for (int nt = 0; nt < 4; ++nt) {
                const h16* bp = &Bs[buf][nt * 16 + n_ld][kc + b_koff];
                v8h fbl = *(const v8h*)(bp);
                v8h fbh = *(const v8h*)(bp + 16);
                v16h bfrag;
#pragma unroll
                for (int i = 0; i < 8; ++i) { bfrag[i] = fbl[i]; bfrag[i + 8] = fbh[i]; }
                acc[nt] = __builtin_amdgcn_wmma_f32_16x16x32_f16(
                    false, afrag, false, bfrag, (short)0, acc[nt], false, false);
            }
        }
        __syncthreads();   // all reads of buf done before it is refilled
    }

    // epilogue: BN scale/shift + ReLU, scatter to NCHW
    const int mrow0 = m_base + wave * 16 + ((lane >= 16) ? 8 : 0);
#pragma unroll
    for (int nt = 0; nt < 4; ++nt) {
        int ncol = n_base + nt * 16 + (lane & 15);
        float sc = scale[ncol], sh = shift[ncol];
        float* op = out + ((size_t)(b * CO + ncol)) * HW;
#pragma unroll
        for (int v = 0; v < 8; ++v) {
            float val = acc[nt][v] * sc + sh;
            op[mrow0 + v] = fmaxf(val, 0.0f);
        }
    }
}

// ------------------------------------------------------------------
extern "C" void kernel_launch(void* const* d_in, const int* in_sizes, int n_in,
                              void* d_out, int out_size, void* d_ws, size_t ws_size,
                              hipStream_t stream) {
    (void)in_sizes; (void)n_in; (void)out_size; (void)ws_size;
    const float* x     = (const float*)d_in[0];
    const float* w_off = (const float*)d_in[1];
    const float* b_off = (const float*)d_in[2];
    const float* w     = (const float*)d_in[3];
    const float* bias  = (const float*)d_in[4];
    const float* gamma = (const float*)d_in[5];
    const float* beta  = (const float*)d_in[6];
    const float* rmean = (const float*)d_in[7];
    const float* rvar  = (const float*)d_in[8];
    float* out = (float*)d_out;

    char* ws = (char*)d_ws;
    float*  scale = (float*)(ws + OFF_SCALE);
    float*  shift = (float*)(ws + OFF_SHIFT);
    h16*    Bw    = (h16*)  (ws + OFF_BW);
    int4*   widx  = (int4*) (ws + OFF_WIDX);
    float4* wwgt  = (float4*)(ws + OFF_WWGT);
    h16*    Amat  = (h16*)  (ws + OFF_A);

    convert_kernel<<<(CO * KDIM + 255) / 256, 256, 0, stream>>>(
        w, bias, gamma, beta, rmean, rvar, Bw, scale, shift);

    for (int b = 0; b < CB; ++b) {
        offset_meta_kernel<<<HW / 256, 256, 0, stream>>>(x, w_off, b_off, b, widx, wwgt);
        sample_kernel<<<HW, 256, 0, stream>>>(x, widx, wwgt, Amat, b);
        gemm_kernel<<<dim3(HW / 128, CO / 64), 256, 0, stream>>>(
            Amat, Bw, scale, shift, out, b);
    }
}
